// EMAVectorQuantizer_37821482009269
// MI455X (gfx1250) — matrix operand order repaired and checked by reading, not compiled, not run on hardware
//
#include <hip/hip_runtime.h>
#include <hip/hip_bf16.h>

// ---------------------------------------------------------------------------
// EMA Vector Quantizer forward for MI455X (gfx1250).
// B=8, R=16, C=512, S=512, K=64 ; M = B*R*C = 65536 rows.
// Core: [65536 x 64] x [64 x 512] GEMM on v_wmma_f32_16x16x32_bf16 with
// bf16 hi/lo fp32-split (3 WMMAs per K=32 chunk), then argmin / gather.
// ---------------------------------------------------------------------------

#define M_TOTAL   65536
#define S_CODES   512
#define K_DIM     64
#define N_TILES   32          // 512 / 16
#define OUT_HAT   0
#define OUT_Z     4194304
#define OUT_LC    4259840
#define OUT_LB    4259841
#define OUT_E     4259842

// ---- workspace layout (bytes) ----
#define WS_CNORM  0                         // 512 f32
#define WS_BHI    2048                      // 32*2*32*8 u32 = 65536 B
#define WS_BLO    (WS_BHI + 65536)
#define WS_ACC    (WS_BLO + 65536)          // 1 f32

typedef __attribute__((ext_vector_type(16))) __bf16        v16bf;
typedef __attribute__((ext_vector_type(8)))  float         v8f;
typedef __attribute__((ext_vector_type(8)))  unsigned int  v8u;

__device__ __forceinline__ unsigned f2bf(float f) {
  unsigned u = __float_as_uint(f);
  return (u + 0x7FFFu + ((u >> 16) & 1u)) >> 16;   // round-to-nearest-even
}
__device__ __forceinline__ float bf2f(unsigned h) {
  return __uint_as_float(h << 16);
}
__device__ __forceinline__ v16bf as_bf(v8u u) {
  union { v8u u; v16bf b; } c; c.u = u; return c.b;
}

// ---------------------------------------------------------------------------
// Prep: cnorm2[s] = ||c_s||^2 ; pack codebook into WMMA bf16 B-fragments.
// B-matrix (32x16 bf16) per-lane layout: lane = (kk/16)*16 + N holds column
// N, K-halves packed 2-per-VGPR: VGPR j = { K=16*(lane/16)+2j , +1 }.
// ---------------------------------------------------------------------------
__global__ void vq_prep(const float* __restrict__ c_sum,
                        const float* __restrict__ c_count,
                        float* __restrict__ cnorm2,
                        unsigned int* __restrict__ bhi,
                        unsigned int* __restrict__ blo,
                        float* __restrict__ acc) {
  const int s = blockIdx.x;     // 0..511
  const int k = threadIdx.x;    // 0..63
  const float inv = 1.0f / fmaxf(c_count[s], 0.01f);
  const float v = c_sum[s * K_DIM + k] * inv;

  __shared__ float red[64];
  red[k] = v * v;
  __syncthreads();
  for (int o = 32; o > 0; o >>= 1) {
    if (k < o) red[k] += red[k + o];
    __syncthreads();
  }
  if (k == 0) cnorm2[s] = red[0];

  if ((k & 1) == 0) {
    const float v1 = c_sum[s * K_DIM + k + 1] * inv;
    const unsigned h0 = f2bf(v),              h1 = f2bf(v1);
    const unsigned l0 = f2bf(v  - bf2f(h0)),  l1 = f2bf(v1 - bf2f(h1));
    const int nt = s >> 4, N = s & 15;
    const int kc = k >> 5, kk = k & 31;
    const int lane = ((kk >> 4) << 4) + N;
    const int j = (kk & 15) >> 1;
    const int idx = (((nt * 2 + kc) * 32 + lane) << 3) + j;
    bhi[idx] = h0 | (h1 << 16);
    blo[idx] = l0 | (l1 << 16);
  }
  if (s == 0 && k == 0) *acc = 0.0f;
}

// ---------------------------------------------------------------------------
// Main: one block = 16-row M tile x all 512 codes. 8 waves, 4 N-tiles/wave,
// 6 bf16 WMMAs per N-tile (hi*hi + hi*lo + lo*hi over two K=32 chunks).
// ---------------------------------------------------------------------------
__global__ void __launch_bounds__(256)
vq_main(const float* __restrict__ vecs,
        const float* __restrict__ c_sum,
        const float* __restrict__ c_count,
        const float* __restrict__ cnorm2,
        const unsigned int* __restrict__ bhi,
        const unsigned int* __restrict__ blo,
        float* __restrict__ out_hat,
        float* __restrict__ out_z,
        float* __restrict__ out_e,
        float* __restrict__ acc) {
  __shared__ float As[16][K_DIM];       // 4 KiB vec tile
  __shared__ float sdist[16][S_CODES];  // 32 KiB distances
  __shared__ float vn[16];
  __shared__ float pmin[16][16];
  __shared__ int   pidx[16][16];
  __shared__ int   zsh[16];
  __shared__ float sinv[16];
  __shared__ float esh[16];

  const int tid  = threadIdx.x;
  const int lane = tid & 31;
  const int wave = tid >> 5;
  const int m0   = blockIdx.x * 16;

  // stage the 16x64 fp32 A tile
  const float* vbase = vecs + (size_t)m0 * K_DIM;
  for (int i = tid; i < 16 * K_DIM; i += 256) (&As[0][0])[i] = vbase[i];
  __syncthreads();

  // row norms
  if (tid < 16) {
    float s = 0.0f;
    for (int k = 0; k < K_DIM; ++k) { const float x = As[tid][k]; s += x * x; }
    vn[tid] = s;
  }

  // build bf16 hi/lo A fragments (contiguous k order, same as the B packing):
  // lane<16 -> M=lane, K = {0..15}; lane>=16 -> M=lane-16, K={16..31}
  const int M = lane & 15;
  v8u ahi[2], alo[2];
  for (int kc = 0; kc < 2; ++kc) {
    for (int j = 0; j < 8; ++j) {
      const int kb = kc * 32 + ((lane >> 4) << 4) + (j << 1);
      const float x0 = As[M][kb], x1 = As[M][kb + 1];
      const unsigned h0 = f2bf(x0),             h1 = f2bf(x1);
      const unsigned l0 = f2bf(x0 - bf2f(h0)),  l1 = f2bf(x1 - bf2f(h1));
      ahi[kc][j] = h0 | (h1 << 16);
      alo[kc][j] = l0 | (l1 << 16);
    }
  }
  __syncthreads();   // vn + fragments ready

  // each wave computes 4 of the 32 N-tiles
  for (int i = 0; i < 4; ++i) {
    const int nt = wave * 4 + i;
    v8f accv = {};
    for (int kc = 0; kc < 2; ++kc) {
      const v8u bh = *(const v8u*)(bhi + (((nt * 2 + kc) * 32 + lane) << 3));
      const v8u bl = *(const v8u*)(blo + (((nt * 2 + kc) * 32 + lane) << 3));
      accv = __builtin_amdgcn_wmma_f32_16x16x32_bf16(
          false, as_bf(ahi[kc]), false, as_bf(bh), (short)0, accv, false, false);
      accv = __builtin_amdgcn_wmma_f32_16x16x32_bf16(
          false, as_bf(ahi[kc]), false, as_bf(bl), (short)0, accv, false, false);
      accv = __builtin_amdgcn_wmma_f32_16x16x32_bf16(
          false, as_bf(alo[kc]), false, as_bf(bh), (short)0, accv, false, false);
    }
    // C layout: lane<16 -> N=lane, M=r ; lane>=16 -> N=lane-16, M=8+r
    const int N = nt * 16 + (lane & 15);
    const int mb = (lane >> 4) << 3;
    const float cn = cnorm2[N];
    for (int r = 0; r < 8; ++r) {
      const int Mr = mb + r;
      sdist[Mr][N] = vn[Mr] - 2.0f * accv[r] + cn;
    }
  }
  __syncthreads();

  // block-wide argmin over 512 codes per row (first-min tie semantics)
  {
    const int row = tid >> 4, p = tid & 15, c0 = p * 32;
    float mn = 3.402823466e38f; int mi = c0;
    for (int c = 0; c < 32; ++c) {
      const float d = sdist[row][c0 + c];
      if (d < mn) { mn = d; mi = c0 + c; }
    }
    pmin[row][p] = mn; pidx[row][p] = mi;
  }
  __syncthreads();
  if (tid < 16) {
    float mn = pmin[tid][0]; int mi = pidx[tid][0];
    for (int p = 1; p < 16; ++p) {
      const float d = pmin[tid][p];
      if (d < mn) { mn = d; mi = pidx[tid][p]; }
    }
    zsh[tid]  = mi;
    sinv[tid] = 1.0f / fmaxf(c_count[mi], 0.01f);
    const float e = fmaxf(mn, 0.0f);
    esh[tid] = e;
    out_z[m0 + tid] = (float)mi;
    out_e[m0 + tid] = e;
  }
  __syncthreads();
  if (tid == 0) {
    float s = 0.0f;
    for (int i = 0; i < 16; ++i) s += esh[i];
    atomicAdd(acc, s);
  }

  // vecs_hat = c[z] gathered from c_sum * (1/clip(count))  (L2-resident)
  {
    const int row = tid >> 4;
    const int k0  = (tid & 15) * 4;
    const float4 cw = *(const float4*)(c_sum + (size_t)zsh[row] * K_DIM + k0);
    const float iv = sinv[row];
    float4 o; o.x = cw.x * iv; o.y = cw.y * iv; o.z = cw.z * iv; o.w = cw.w * iv;
    *(float4*)(out_hat + (size_t)(m0 + row) * K_DIM + k0) = o;
  }
}

__global__ void vq_fin(const float* __restrict__ acc,
                       float* __restrict__ out_lc,
                       float* __restrict__ out_lb) {
  *out_lc = *acc * (1.0f / (float)M_TOTAL);
  *out_lb = 0.0f;
}

// ---------------------------------------------------------------------------
extern "C" void kernel_launch(void* const* d_in, const int* in_sizes, int n_in,
                              void* d_out, int out_size, void* d_ws, size_t ws_size,
                              hipStream_t stream) {
  const float* vecs    = (const float*)d_in[0];
  const float* c_sum   = (const float*)d_in[1];
  const float* c_count = (const float*)d_in[2];
  float* out = (float*)d_out;
  char*  ws  = (char*)d_ws;

  float*        cnorm2 = (float*)(ws + WS_CNORM);
  unsigned int* bhi    = (unsigned int*)(ws + WS_BHI);
  unsigned int* blo    = (unsigned int*)(ws + WS_BLO);
  float*        acc    = (float*)(ws + WS_ACC);

  vq_prep<<<S_CODES, K_DIM, 0, stream>>>(c_sum, c_count, cnorm2, bhi, blo, acc);
  vq_main<<<M_TOTAL / 16, 256, 0, stream>>>(vecs, c_sum, c_count, cnorm2, bhi, blo,
                                            out + OUT_HAT, out + OUT_Z, out + OUT_E, acc);
  vq_fin<<<1, 1, 0, stream>>>(acc, out + OUT_LC, out + OUT_LB);
}
